// TOGL_72413148611005
// MI455X (gfx1250) — matrix operand order repaired and checked by reading, not compiled, not run on hardware
//
#include <hip/hip_runtime.h>
#include <hip/hip_bf16.h>
#include <stdint.h>

typedef __attribute__((ext_vector_type(16))) __bf16 v16bf;
typedef __attribute__((ext_vector_type(8)))  float  v8f;

#define NFEAT 512   // n_feat (input / output features)
#define HID   512   // filtration MLP hidden
#define NFILT 8     // number of filtrations
#define OUTD  64    // deepset width
#define BN_EPS 1e-5f

__device__ __forceinline__ uint16_t f32_to_bf16(float f) {
    union { float f; uint32_t u; } c; c.f = f;
    uint32_t u = c.u;
    return (uint16_t)((u + 0x7FFFu + ((u >> 16) & 1u)) >> 16);
}
__device__ __forceinline__ float bf16_to_f32(uint16_t h) {
    union { uint32_t u; float f; } c; c.u = ((uint32_t)h) << 16;
    return c.f;
}

// ---------------------------------------------------------------- converts
__global__ void cvt_bf16_kernel(const float* __restrict__ in,
                                uint16_t* __restrict__ out, unsigned n) {
    unsigned i = blockIdx.x * blockDim.x + threadIdx.x;
    if (i < n) out[i] = f32_to_bf16(in[i]);
}

// W1 [K=512][N=512] row-major f32 -> w1t [N][K] bf16 (one-time, 512 KB)
__global__ void cvt_w1t_kernel(const float* __restrict__ W1,
                               uint16_t* __restrict__ w1t) {
    unsigned gid = blockIdx.x * blockDim.x + threadIdx.x;
    if (gid >= (unsigned)NFEAT * HID) return;
    unsigned n = gid / HID, k = gid % HID;      // coalesced writes along k
    w1t[gid] = f32_to_bf16(W1[(size_t)k * NFEAT + n]);
}

__global__ void zero_kernel(float* p, int n) {
    int i = blockIdx.x * blockDim.x + threadIdx.x;
    if (i < n) p[i] = 0.f;
}

// ---------------------------------------------------------------- big GEMM
// hidden = relu(xb[M,512] @ W1 + b1), bf16 in / f32 acc / bf16 out.
// Block tile 128x64, 8 waves (4x2), each wave 32x32 = 2x2 WMMA 16x16x32 bf16.
// B operand (w1t) is pre-transposed to [N][K] so both tiles stage with b128.
__global__ __launch_bounds__(256)
void gemm1_wmma(const uint16_t* __restrict__ xb, const uint16_t* __restrict__ w1t,
                const float* __restrict__ b1, uint16_t* __restrict__ hb, int M) {
    __shared__ __align__(16) uint16_t As[128 * 32];   // [row][k]
    __shared__ __align__(16) uint16_t Bs[64 * 32];    // [col][k]

    const int tid  = threadIdx.x;
    const int lane = tid & 31;
    const int wave = tid >> 5;
    const int wm   = wave & 3;     // M strip (0..3) of 32 rows
    const int wn   = wave >> 2;    // N strip (0..1) of 32 cols
    const int half = lane >> 4;
    const int lr   = lane & 15;
    const int mBase = blockIdx.x * 128;
    const int nBase = blockIdx.y * 64;

    v8f acc[2][2];
    #pragma unroll
    for (int i = 0; i < 2; ++i)
        #pragma unroll
        for (int j = 0; j < 2; ++j)
            acc[i][j] = (v8f){0.f,0.f,0.f,0.f,0.f,0.f,0.f,0.f};

    union Frag { v16bf v; uint4 u[2]; };

    for (int k0 = 0; k0 < HID; k0 += 32) {
        // --- stage A tile 128x32 (16 bf16 per thread, two b128 loads)
        #pragma unroll
        for (int c = 0; c < 2; ++c) {
            int l   = tid * 16 + c * 8;
            int row = l >> 5, kk = l & 31;
            *(uint4*)(As + row * 32 + kk) =
                *(const uint4*)(xb + (size_t)(mBase + row) * HID + k0 + kk);
        }
        // --- stage B tile [64 cols][32 k] with a single b128 per thread
        {
            int n = tid & 63, kb = (tid >> 6) * 8;
            *(uint4*)(Bs + n * 32 + kb) =
                *(const uint4*)(w1t + (size_t)(nBase + n) * HID + k0 + kb);
        }
        if (k0 + 32 < HID)  // prefetch next A K-slab (global_prefetch_b8)
            __builtin_prefetch(xb + (size_t)(mBase + (tid >> 1)) * HID + k0 + 32, 0, 1);
        __syncthreads();

        Frag a[2], b[2];
        #pragma unroll
        for (int t = 0; t < 2; ++t) {
            int row = wm * 32 + t * 16 + lr;
            a[t].u[0] = *(const uint4*)(As + row * 32 + half * 8);        // K 0..7 / 8..15
            a[t].u[1] = *(const uint4*)(As + row * 32 + 16 + half * 8);   // K 16..23 / 24..31
            int col = wn * 32 + t * 16 + lr;
            b[t].u[0] = *(const uint4*)(Bs + col * 32 + half * 8);        // K 0..7 / 8..15
            b[t].u[1] = *(const uint4*)(Bs + col * 32 + 16 + half * 8);   // K 16..23 / 24..31
        }
        #pragma unroll
        for (int i = 0; i < 2; ++i)
            #pragma unroll
            for (int j = 0; j < 2; ++j)
                acc[i][j] = __builtin_amdgcn_wmma_f32_16x16x32_bf16(
                    false, a[i].v, false, b[j].v, (short)0, acc[i][j], false, false);
        __syncthreads();
    }

    // --- bias + relu + bf16 store. C layout: lane = (m>=8)*16 + n, vgpr = m%8
    #pragma unroll
    for (int i = 0; i < 2; ++i) {
        int mB = mBase + wm * 32 + i * 16 + half * 8;
        #pragma unroll
        for (int j = 0; j < 2; ++j) {
            int n = nBase + wn * 32 + j * 16 + lr;
            float bias = b1[n];
            #pragma unroll
            for (int v = 0; v < 8; ++v) {
                float val = fmaxf(acc[i][j][v] + bias, 0.f);
                hb[(size_t)(mB + v) * HID + n] = f32_to_bf16(val);
            }
        }
    }
}

// ---------------------------------------------------------------- fv = hidden@W2 + b2
__global__ void fv_kernel(const uint16_t* __restrict__ hb, const float* __restrict__ W2,
                          const float* __restrict__ b2, float* __restrict__ fv, int Ntot) {
    int gid = blockIdx.x * blockDim.x + threadIdx.x;
    if (gid >= Ntot * NFILT) return;
    int n = gid / NFILT, k = gid % NFILT;
    float acc = b2[k];
    const uint16_t* row = hb + (size_t)n * HID;
    for (int j = 0; j < HID; ++j)
        acc += bf16_to_f32(row[j]) * W2[j * NFILT + k];
    fv[gid] = acc;
}

// ---------------------------------------------------------------- persistence pairing
// One workgroup per (graph, filtration): bitonic sort of edges in LDS, then
// serial elder-rule union-find on lane 0.
__global__ __launch_bounds__(256)
void persistence_kernel(const float* __restrict__ fv, const int* __restrict__ eu,
                        const int* __restrict__ ev, int* __restrict__ death,
                        int Ntot, int npg, int epg, int E2) {
    __shared__ unsigned long long keys[2048];
    __shared__ int parent[1024];
    __shared__ int birth[1024];
    __shared__ float fvl[1024];
    __shared__ unsigned short ul[2048], vlidx[2048];

    const int f = blockIdx.x & (NFILT - 1);
    const int g = blockIdx.x >> 3;
    const int vbase = g * npg, ebase = g * epg;
    const int tid = threadIdx.x;

    for (int i = tid; i < npg; i += 256) {
        fvl[i] = fv[(size_t)(vbase + i) * NFILT + f];
        parent[i] = i;
        birth[i] = i;
        death[(size_t)f * Ntot + vbase + i] = vbase + i;  // essential default
    }
    __syncthreads();

    for (int e = tid; e < E2; e += 256) {
        if (e < epg) {
            int a = eu[ebase + e] - vbase, b = ev[ebase + e] - vbase;
            ul[e] = (unsigned short)a;
            vlidx[e] = (unsigned short)b;
            float fe = fmaxf(fvl[a], fvl[b]);
            union { float f; uint32_t u; } c; c.f = fe;
            uint32_t s = (c.u & 0x80000000u) ? ~c.u : (c.u | 0x80000000u);
            keys[e] = ((unsigned long long)s << 32) | (unsigned)e;  // stable tie-break
        } else {
            keys[e] = ~0ull;
        }
    }
    __syncthreads();

    for (int ks = 2; ks <= E2; ks <<= 1) {
        for (int j = ks >> 1; j > 0; j >>= 1) {
            for (int i = tid; i < E2; i += 256) {
                int ixj = i ^ j;
                if (ixj > i) {
                    bool up = (i & ks) == 0;
                    unsigned long long xk = keys[i], yk = keys[ixj];
                    if ((xk > yk) == up) { keys[i] = yk; keys[ixj] = xk; }
                }
            }
            __syncthreads();
        }
    }

    if (tid == 0) {
        for (int s = 0; s < epg; ++s) {
            int e = (int)(keys[s] & 0xFFFFFFFFu);
            int a = ul[e], b = vlidx[e];
            int ra = a; while (parent[ra] != ra) ra = parent[ra];
            int rb = b; while (parent[rb] != rb) rb = parent[rb];
            int x = a; while (parent[x] != ra) { int nx = parent[x]; parent[x] = ra; x = nx; }
            x = b;     while (parent[x] != rb) { int nx = parent[x]; parent[x] = rb; x = nx; }
            if (ra == rb) continue;
            int ba = birth[ra], bb = birth[rb];
            float fa = fvl[ba], fb = fvl[bb];
            bool le = (fa < fb) || (fa == fb && ba <= bb);
            int elder = le ? ba : bb;
            int young = le ? bb : ba;
            float fea = fvl[a], feb = fvl[b];
            int w = ((fea > feb) || (fea == feb && a >= b)) ? a : b;
            death[(size_t)f * Ntot + vbase + young] = vbase + w;
            parent[ra] = rb;
            birth[rb] = elder;
        }
    }
}

// ---------------------------------------------------------------- pd -> Linear(16,64)+ReLU
__global__ void pd_linear_kernel(const float* __restrict__ fv, const int* __restrict__ death,
                                 const float* __restrict__ W_in, const float* __restrict__ b_in,
                                 float* __restrict__ h1, int Ntot) {
    int gid = blockIdx.x * blockDim.x + threadIdx.x;
    if (gid >= Ntot * OUTD) return;
    int n = gid / OUTD, j = gid % OUTD;
    float acc = b_in[j];
    #pragma unroll
    for (int k = 0; k < NFILT; ++k) {
        float bth = fv[(size_t)n * NFILT + k];
        int di = death[(size_t)k * Ntot + n];
        float dth = fv[(size_t)di * NFILT + k];
        acc += bth * W_in[(2 * k) * OUTD + j] + dth * W_in[(2 * k + 1) * OUTD + j];
    }
    h1[gid] = fmaxf(acc, 0.f);
}

// ---------------------------------------------------------------- per-graph mean @ lW
__global__ __launch_bounds__(64)
void lambda1_kernel(const float* __restrict__ h, const float* __restrict__ lW,
                    float* __restrict__ xm, int npg) {
    __shared__ float mean[OUTD];
    int g = blockIdx.x, j = threadIdx.x;
    float s = 0.f;
    for (int r = 0; r < npg; ++r) s += h[((size_t)g * npg + r) * OUTD + j];
    mean[j] = s / (float)npg;
    __syncthreads();
    float acc = 0.f;
    for (int i = 0; i < OUTD; ++i) acc += mean[i] * lW[i * OUTD + j];
    xm[g * OUTD + j] = acc;
}

__global__ void deepset1_kernel(const float* __restrict__ h1, const float* __restrict__ g1W,
                                const float* __restrict__ g1b, const float* __restrict__ xm1,
                                const int* __restrict__ batch, float* __restrict__ h2, int Ntot) {
    int gid = blockIdx.x * blockDim.x + threadIdx.x;
    if (gid >= Ntot * OUTD) return;
    int n = gid / OUTD, j = gid % OUTD;
    float acc = g1b[j];
    const float* row = h1 + (size_t)n * OUTD;
    for (int i = 0; i < OUTD; ++i) acc += row[i] * g1W[i * OUTD + j];
    acc -= xm1[batch[n] * OUTD + j];
    h2[gid] = fmaxf(acc, 0.f);
}

__global__ __launch_bounds__(64)
void lambda2_kernel(const float* __restrict__ h2, const float* __restrict__ l2W,
                    float* __restrict__ xm2, int npg) {
    __shared__ float mean[OUTD];
    int g = blockIdx.x, j = threadIdx.x;
    float s = 0.f;
    for (int r = 0; r < npg; ++r) s += h2[((size_t)g * npg + r) * OUTD + j];
    mean[j] = s / (float)npg;
    __syncthreads();
    for (int c = 0; c < NFEAT / OUTD; ++c) {
        int m = c * OUTD + j;
        float acc = 0.f;
        for (int i = 0; i < OUTD; ++i) acc += mean[i] * l2W[i * NFEAT + m];
        xm2[(size_t)g * NFEAT + m] = acc;
    }
}

__global__ __launch_bounds__(256)
void deepset2_kernel(const float* __restrict__ h2, const float* __restrict__ g2W,
                     const float* __restrict__ g2b, const float* __restrict__ xm2,
                     const int* __restrict__ batch, float* __restrict__ rbuf, int Ntot) {
    __shared__ float row[OUTD];
    int n = blockIdx.x;
    int t = threadIdx.x;
    if (t < OUTD) row[t] = h2[(size_t)n * OUTD + t];
    __syncthreads();
    int b = batch[n];
    #pragma unroll
    for (int c = 0; c < 2; ++c) {
        int m = t + c * 256;
        float acc = g2b[m] - xm2[(size_t)b * NFEAT + m];
        for (int i = 0; i < OUTD; ++i) acc += row[i] * g2W[i * NFEAT + m];
        rbuf[(size_t)n * NFEAT + m] = fmaxf(acc, 0.f);  // r = relu(z)
    }
}

// ---------------------------------------------------------------- BN stats + finalize
__global__ __launch_bounds__(256)
void bnstat_kernel(const float* __restrict__ rbuf, float* __restrict__ bnsum,
                   float* __restrict__ bnsq, int rowsPer, int Ntot) {
    int t = threadIdx.x;
    int r0 = blockIdx.x * rowsPer;
    float s0 = 0.f, s1 = 0.f, q0 = 0.f, q1 = 0.f;
    for (int rr = 0; rr < rowsPer; ++rr) {
        int row = r0 + rr;
        if (row >= Ntot) break;
        float a = rbuf[(size_t)row * NFEAT + t];
        float b = rbuf[(size_t)row * NFEAT + t + 256];
        s0 += a; q0 += a * a;
        s1 += b; q1 += b * b;
    }
    atomicAdd(&bnsum[t], s0);       atomicAdd(&bnsq[t], q0);
    atomicAdd(&bnsum[t + 256], s1); atomicAdd(&bnsq[t + 256], q1);
}

__global__ void bn_out_kernel(const float* __restrict__ x, const float* __restrict__ rbuf,
                              const float* __restrict__ bnsum, const float* __restrict__ bnsq,
                              const float* __restrict__ gamma, const float* __restrict__ beta,
                              float* __restrict__ out, int Ntot) {
    int gid = blockIdx.x * blockDim.x + threadIdx.x;
    if (gid >= Ntot * NFEAT) return;   // 16.7M < 2^31
    int m = gid & (NFEAT - 1);
    float inv = 1.f / (float)Ntot;
    float mu = bnsum[m] * inv;
    float var = bnsq[m] * inv - mu * mu;
    out[gid] = x[gid] + gamma[m] * (rbuf[gid] - mu) * rsqrtf(var + BN_EPS) + beta[m];
}

// ================================================================ launch
extern "C" void kernel_launch(void* const* d_in, const int* in_sizes, int n_in,
                              void* d_out, int out_size, void* d_ws, size_t ws_size,
                              hipStream_t stream) {
    const float* x     = (const float*)d_in[0];
    const int*   ei    = (const int*)d_in[1];
    const int*   batch = (const int*)d_in[2];
    const float* W1    = (const float*)d_in[5];
    const float* b1    = (const float*)d_in[6];
    const float* W2    = (const float*)d_in[7];
    const float* b2    = (const float*)d_in[8];
    const float* W_in  = (const float*)d_in[9];
    const float* b_in  = (const float*)d_in[10];
    const float* g1W   = (const float*)d_in[11];
    const float* g1b   = (const float*)d_in[12];
    const float* l1W   = (const float*)d_in[13];
    const float* g2W   = (const float*)d_in[14];
    const float* g2b   = (const float*)d_in[15];
    const float* l2W   = (const float*)d_in[16];
    const float* gamma = (const float*)d_in[17];
    const float* beta  = (const float*)d_in[18];
    float* out = (float*)d_out;

    const int Ntot = in_sizes[0] / NFEAT;     // 32768
    const int E    = in_sizes[1] / 2;         // 65536
    const int B    = in_sizes[3] - 1;         // 32
    const int npg  = Ntot / B;                // 1024
    const int epg  = E / B;                   // 2048
    const int* eu = ei;
    const int* ev = ei + E;

    char* ws = (char*)d_ws;
    size_t off = 0;
    auto alloc = [&](size_t bytes) -> void* {
        void* p = ws + off;
        off += (bytes + 255) & ~(size_t)255;
        return p;
    };
    uint16_t* xb    = (uint16_t*)alloc((size_t)Ntot * NFEAT * 2);
    uint16_t* w1t   = (uint16_t*)alloc((size_t)NFEAT * HID * 2);
    uint16_t* hb    = (uint16_t*)alloc((size_t)Ntot * HID * 2);
    float*    fv    = (float*)alloc((size_t)Ntot * NFILT * 4);
    int*      death = (int*)alloc((size_t)NFILT * Ntot * 4);
    float*    h1    = (float*)alloc((size_t)Ntot * OUTD * 4);
    float*    xm1   = (float*)alloc((size_t)B * OUTD * 4);
    float*    h2    = (float*)alloc((size_t)Ntot * OUTD * 4);
    float*    xm2   = (float*)alloc((size_t)B * NFEAT * 4);
    float*    rbuf  = (float*)alloc((size_t)Ntot * NFEAT * 4);
    float*    bnsum = (float*)alloc(NFEAT * 4);
    float*    bnsq  = (float*)alloc(NFEAT * 4);

    const unsigned nx = (unsigned)Ntot * NFEAT;
    cvt_bf16_kernel<<<(nx + 255) / 256, 256, 0, stream>>>(x, xb, nx);
    cvt_w1t_kernel<<<(NFEAT * HID + 255) / 256, 256, 0, stream>>>(W1, w1t);

    dim3 gdim(Ntot / 128, NFEAT / 64);
    gemm1_wmma<<<gdim, 256, 0, stream>>>(xb, w1t, b1, hb, Ntot);

    fv_kernel<<<(Ntot * NFILT + 255) / 256, 256, 0, stream>>>(hb, W2, b2, fv, Ntot);

    int E2 = 1; while (E2 < epg) E2 <<= 1;    // 2048
    persistence_kernel<<<B * NFILT, 256, 0, stream>>>(fv, eu, ev, death, Ntot, npg, epg, E2);

    pd_linear_kernel<<<(Ntot * OUTD + 255) / 256, 256, 0, stream>>>(fv, death, W_in, b_in, h1, Ntot);

    lambda1_kernel<<<B, 64, 0, stream>>>(h1, l1W, xm1, npg);
    deepset1_kernel<<<(Ntot * OUTD + 255) / 256, 256, 0, stream>>>(h1, g1W, g1b, xm1, batch, h2, Ntot);

    lambda2_kernel<<<B, 64, 0, stream>>>(h2, l2W, xm2, npg);
    deepset2_kernel<<<Ntot, 256, 0, stream>>>(h2, g2W, g2b, xm2, batch, rbuf, Ntot);

    zero_kernel<<<(NFEAT + 255) / 256, 256, 0, stream>>>(bnsum, NFEAT);
    zero_kernel<<<(NFEAT + 255) / 256, 256, 0, stream>>>(bnsq, NFEAT);
    const int rowsPer = 128;
    bnstat_kernel<<<(Ntot + rowsPer - 1) / rowsPer, 256, 0, stream>>>(rbuf, bnsum, bnsq, rowsPer, Ntot);

    bn_out_kernel<<<(nx + 255) / 256, 256, 0, stream>>>(x, rbuf, bnsum, bnsq, gamma, beta, out, Ntot);

    (void)n_in; (void)out_size; (void)ws_size;
}
